// DA_89404039234107
// MI455X (gfx1250) — matrix operand order repaired and checked by reading, not compiled, not run on hardware
//
#include <hip/hip_runtime.h>
#include <hip/hip_fp16.h>

typedef _Float16 h16;
typedef __attribute__((ext_vector_type(16))) _Float16 v16h;
typedef __attribute__((ext_vector_type(8)))  _Float16 v8h;
typedef __attribute__((ext_vector_type(8)))  float    v8f;

#define B_  1024
#define L_  256
#define D_  70
#define H_  70
#define NTOK (2 * B_ * L_)       // 524288 tokens across both sentences
#define CPAD 160                 // padded 2H (K multiple of 32)
#define PPAD 96                  // padded D / H for K (multiple of 32)
#define SPAD 80                  // padded H row in sbuf (16B-aligned rows)

// ---------------- CDNA5 async global->LDS staging (ASYNCcnt path) ----------------
// Per-lane: LDS[lds_addr] = MEM[gaddr] (16 bytes), tracked by ASYNCcnt.
__device__ __forceinline__ void async_b128(unsigned lds_addr, const void* g) {
  asm volatile("global_load_async_to_lds_b128 %0, %1, off"
               :: "v"(lds_addr), "v"(g) : "memory");
}
__device__ __forceinline__ void async_wait0() {
  asm volatile("s_wait_asynccnt 0x0" ::: "memory");
}
__device__ __forceinline__ unsigned lds_addr_of(const void* p) {
  return (unsigned)(unsigned long long)p;   // low 32 bits = LDS byte offset
}

// ---------------- WMMA helpers (CDNA5 16x16x32 f16 -> f32) ----------------

__device__ __forceinline__ v8f wmma16(v16h a, v16h b, v8f c) {
  return __builtin_amdgcn_wmma_f32_16x16x32_f16(false, a, false, b,
                                                (short)0, c, false, false);
}

// A fragment (16x32, M x K). X points at tile origin (row0, k0), row-major, ld in halves.
// lane<16: M=lane, K={0..7,16..23}; lane>=16: M=lane-16, K={8..15,24..31}
__device__ __forceinline__ v16h load_a16(const h16* X, int ld, int lane) {
  int m  = lane & 15;
  int kb = (lane >> 4) << 3;
  const h16* p = X + m * ld;
  v16h a;
#pragma unroll
  for (int t = 0; t < 8; ++t) { a[t] = p[kb + t]; a[8 + t] = p[16 + kb + t]; }
  return a;
}

// B fragment for D = A * W^T : B[k][n] = W[n][k], W row-major, ptr at W[n0][k0].
__device__ __forceinline__ v16h load_bT(const h16* W, int ld, int lane) {
  int n  = lane & 15;
  int ko = (lane >> 4) << 3;
  const h16* p = W + n * ld + ko;
  v16h b;
#pragma unroll
  for (int t = 0; t < 8; ++t) { b[t] = p[t]; b[8 + t] = p[16 + t]; }
  return b;
}

// B fragment for D = A * P : B[k][n] = P[k][n], P row-major, ptr at P[k0][n0].
__device__ __forceinline__ v16h load_bN(const h16* P, int ld, int lane) {
  int n  = lane & 15;
  int ko = (lane >> 4) << 3;
  const h16* p = P + ko * ld + n;
  v16h b;
#pragma unroll
  for (int t = 0; t < 8; ++t) { b[t] = p[t * ld]; b[8 + t] = p[(16 + t) * ld]; }
  return b;
}

// ---------------- Kernel 0: weight prep (f32 -> f16, zero-padded) ----------------

__global__ void k_prep(const float* __restrict__ Wp, const float* __restrict__ Wc1,
                       const float* __restrict__ Wc2,
                       h16* wWp, h16* wWc1, h16* wWc2) {
  int i = blockIdx.x * 256 + threadIdx.x;
  if (i < 80 * PPAD) {              // Wp: [70][70] -> [80][96]
    int r = i / PPAD, c = i % PPAD;
    wWp[i] = (r < H_ && c < D_) ? (h16)Wp[r * D_ + c] : (h16)0.f;
  }
  if (i < 144 * CPAD) {             // Wc1: [140][140] -> [144][160]
    int r = i / CPAD, c = i % CPAD;
    wWc1[i] = (r < 140 && c < 140) ? (h16)Wc1[r * 140 + c] : (h16)0.f;
  }
  if (i < 80 * CPAD) {              // Wc2: [70][140] -> [80][160]
    int r = i / CPAD, c = i % CPAD;
    wWc2[i] = (r < 70 && c < 140) ? (h16)Wc2[r * 140 + c] : (h16)0.f;
  }
}

// ---------------- Kernel 1: embedding gather + projection GEMM + ReLU ----------------

__global__ __launch_bounds__(256) void k_embed_proj(
    const int* __restrict__ sen1, const int* __restrict__ sen2,
    const float* __restrict__ emb, const h16* __restrict__ wWp,
    const float* __restrict__ bp, h16* __restrict__ cbuf) {
  __shared__ h16 sWp[80 * PPAD];
  __shared__ h16 sA[8][16 * PPAD];
  const int tid = threadIdx.x, lane = tid & 31, w = tid >> 5;

  // Async-stage Wp into LDS (960 x b128).
  {
    unsigned dst = lds_addr_of(sWp);
    for (int i = tid; i < (80 * PPAD) / 8; i += 256)
      async_b128(dst + i * 16u, wWp + i * 8);
  }

  const long tile = (long)blockIdx.x * 8 + w;   // one 16-token tile per wave
  const long g0   = tile * 16;

  // Gather 16 embedding rows (f32 -> f16) into the wave's A tile, zero-pad K.
  for (int i = lane; i < 16 * PPAD; i += 32) {
    int r = i / PPAD, c = i % PPAD;
    long g   = g0 + r;
    int  sen = (g >= (long)B_ * L_);
    long t   = g - (long)sen * B_ * L_;
    int  idx = sen ? sen2[t] : sen1[t];
    const float* erow = emb + (long)idx * D_;
    if (c == 0) __builtin_prefetch(erow, 0, 1);   // global_prefetch_b8
    sA[w][i] = (c < D_) ? (h16)erow[c] : (h16)0.f;
  }
  async_wait0();
  __syncthreads();

  const h16* A = sA[w];
  const int n = (lane & 15), mb = (lane >> 4) << 3;
  for (int nt = 0; nt < 5; ++nt) {               // N = 70 -> 5 tiles of 16
    v8f acc = {};
#pragma unroll
    for (int kc = 0; kc < 3; ++kc) {             // K = 96
      v16h af = load_a16(A + kc * 32, PPAD, lane);
      v16h bf = load_bT(sWp + (nt * 16) * PPAD + kc * 32, PPAD, lane);
      acc = wmma16(af, bf, acc);
    }
    int col = nt * 16 + n;
    if (col < H_) {
      float bias = bp[col];
#pragma unroll
      for (int v = 0; v < 8; ++v) {
        float x = acc[v] + bias;
        if (x < 0.f) x = 0.f;
        cbuf[(g0 + v + mb) * CPAD + col] = (h16)x;
      }
    }
  }
  // zero cols [70,160) (ctx slot + K padding)
  for (int i = lane; i < 16 * (CPAD - H_); i += 32) {
    int r = i / (CPAD - H_), c = H_ + i % (CPAD - H_);
    cbuf[(g0 + r) * CPAD + c] = (h16)0.f;
  }
}

// ---------------- Kernel 2: per-batch attention, fully in LDS ----------------

__global__ __launch_bounds__(512) void k_attn(h16* __restrict__ cbuf) {
  extern __shared__ char smem[];
  h16*   p1s  = (h16*)smem;                 // [256][96]  48KB
  h16*   p2s  = p1s + 256 * PPAD;           // [256][96]  48KB
  h16*   S    = p2s + 256 * PPAD;           // [256][256] 128KB
  float* rmax = (float*)(S + 256 * 256);
  float* rinv = rmax + 256;
  float* cmax = rinv + 256;
  float* cinv = cmax + 256;

  const int b = blockIdx.x, tid = threadIdx.x, lane = tid & 31, w = tid >> 5;

  // Async-stage p1/p2 (first 96 halves of each 160-half cbuf row): 12 b128 per row.
  {
    const h16* s1 = cbuf + ((long)0 * B_ * L_ + (long)b * L_) * CPAD;
    const h16* s2 = cbuf + ((long)1 * B_ * L_ + (long)b * L_) * CPAD;
    unsigned d1 = lds_addr_of(p1s), d2 = lds_addr_of(p2s);
    for (int i = tid; i < 256 * 12; i += 512) {
      int r = i / 12, q = i % 12;
      async_b128(d1 + i * 16u, s1 + (r * CPAD + q * 8));
      async_b128(d2 + i * 16u, s2 + (r * CPAD + q * 8));
    }
  }
  async_wait0();
  __syncthreads();

  // Scores: wave w owns row-tile w, sweeps 16 col-tiles.
  {
    const int ti = w, n = lane & 15, mb = (lane >> 4) << 3;
    for (int tj = 0; tj < 16; ++tj) {
      v8f acc = {};
#pragma unroll
      for (int kc = 0; kc < 3; ++kc) {
        v16h af = load_a16(p1s + ti * 16 * PPAD + kc * 32, PPAD, lane);
        v16h bf = load_bT (p2s + tj * 16 * PPAD + kc * 32, PPAD, lane);
        acc = wmma16(af, bf, acc);
      }
#pragma unroll
      for (int v = 0; v < 8; ++v)
        S[(ti * 16 + v + mb) * 256 + tj * 16 + n] = (h16)acc[v];
    }
  }
  __syncthreads();

  // Softmax stats: threads 0..255 -> rows (b128-vectorized), 256..511 -> cols.
  if (tid < 256) {
    int r = tid;
    const v8h* row = (const v8h*)(S + r * 256);
    float m = -1e30f;
    for (int k = 0; k < 32; ++k) {
      v8h v = row[k];
#pragma unroll
      for (int t = 0; t < 8; ++t) m = fmaxf(m, (float)v[t]);
    }
    float s = 0.f;
    for (int k = 0; k < 32; ++k) {
      v8h v = row[k];
#pragma unroll
      for (int t = 0; t < 8; ++t) s += __expf((float)v[t] - m);
    }
    rmax[r] = m; rinv[r] = 1.f / s;
  } else {
    int c = tid - 256;
    float m = -1e30f;
    for (int k = 0; k < 256; ++k) m = fmaxf(m, (float)S[k * 256 + c]);
    float s = 0.f;
    for (int k = 0; k < 256; ++k) s += __expf((float)S[k * 256 + c] - m);
    cmax[c] = m; cinv[c] = 1.f / s;
  }
  __syncthreads();

  // Context GEMMs: 2 x 16 row-tiles x 5 N-tiles = 160 tiles over 16 waves.
  const int n = lane & 15, mb = (lane >> 4) << 3, kb = (lane >> 4) << 3;
  for (int q = w; q < 160; q += 16) {
    const int which = q / 80;          // 0: ctx1 (rows of sen1), 1: ctx2 (rows of sen2)
    const int rq = q % 80;
    const int ti = rq / 5, nt = rq % 5;
    const int row = ti * 16 + n;       // this lane's A row
    const float rm = which ? cmax[row] : rmax[row];
    const float ri = which ? cinv[row] : rinv[row];
    v8f acc = {};
    for (int kc = 0; kc < 8; ++kc) {   // K = 256
      const int k0 = kc * 32;
      v16h af;
#pragma unroll
      for (int t = 0; t < 8; ++t) {    // softmax-normalized A frag on the fly
        int k1 = k0 + kb + t, k2 = k0 + 16 + kb + t;
        float x1 = which ? (float)S[k1 * 256 + row] : (float)S[row * 256 + k1];
        float x2 = which ? (float)S[k2 * 256 + row] : (float)S[row * 256 + k2];
        af[t]     = (h16)(__expf(x1 - rm) * ri);
        af[8 + t] = (h16)(__expf(x2 - rm) * ri);
      }
      const h16* P = which ? p1s : p2s;
      v16h bf = load_bN(P + k0 * PPAD + nt * 16, PPAD, lane);
      acc = wmma16(af, bf, acc);
    }
    const int col = nt * 16 + n;
    if (col < H_) {
      long base = ((long)which * B_ * L_ + (long)b * L_ + ti * 16);
#pragma unroll
      for (int v = 0; v < 8; ++v)
        cbuf[(base + v + mb) * CPAD + H_ + col] = (h16)acc[v];
    }
  }
}

// ---------------- Kernel 3: combined MLP (140->140->70, both ReLU) ----------------

__global__ __launch_bounds__(256) void k_mlp(const h16* __restrict__ cbuf,
                                             const h16* __restrict__ wWc1,
                                             const h16* __restrict__ wWc2,
                                             const float* __restrict__ bc1,
                                             const float* __restrict__ bc2,
                                             h16* __restrict__ sbuf) {
  extern __shared__ char smem[];
  h16* sW1 = (h16*)smem;                 // [144][160]
  h16* sW2 = sW1 + 144 * CPAD;           // [80][160]
  h16* sC  = sW2 + 80 * CPAD;            // 8 x [16][160]
  h16* sH  = sC + 8 * 16 * CPAD;         // 8 x [16][160]
  const int tid = threadIdx.x, lane = tid & 31, w = tid >> 5;

  // Async-stage both weight matrices and this wave's c-tile.
  {
    unsigned dW1 = lds_addr_of(sW1);
    for (int i = tid; i < (144 * CPAD) / 8; i += 256)
      async_b128(dW1 + i * 16u, wWc1 + i * 8);
    unsigned dW2 = lds_addr_of(sW2);
    for (int i = tid; i < (80 * CPAD) / 8; i += 256)
      async_b128(dW2 + i * 16u, wWc2 + i * 8);
  }
  const long g0 = ((long)blockIdx.x * 8 + w) * 16;
  h16* C  = sC + w * 16 * CPAD;
  h16* Hh = sH + w * 16 * CPAD;
  {
    const h16* src = cbuf + g0 * CPAD;     // 16 rows x 160 halves = 320 b128
    unsigned dC = lds_addr_of(C);
    for (int i = lane; i < 320; i += 32)
      async_b128(dC + i * 16u, src + i * 8);
  }
  async_wait0();
  __syncthreads();

  const int n = lane & 15, mb = (lane >> 4) << 3;
  // GEMM1: [16x160] x [160x144], bias+ReLU into Hh (zero-padded for next K).
  for (int nt = 0; nt < 9; ++nt) {
    v8f acc = {};
#pragma unroll
    for (int kc = 0; kc < 5; ++kc) {
      v16h af = load_a16(C + kc * 32, CPAD, lane);
      v16h bf = load_bT(sW1 + nt * 16 * CPAD + kc * 32, CPAD, lane);
      acc = wmma16(af, bf, acc);
    }
    int col = nt * 16 + n;
    float bias = (col < 140) ? bc1[col] : 0.f;
#pragma unroll
    for (int v = 0; v < 8; ++v) {
      float x = acc[v] + bias;
      if (x < 0.f || col >= 140) x = 0.f;
      Hh[(v + mb) * CPAD + col] = (h16)x;
    }
  }
  for (int i = lane; i < 16 * 16; i += 32)      // zero cols 144..159
    Hh[(i >> 4) * CPAD + 144 + (i & 15)] = (h16)0.f;

  // GEMM2: [16x160] x [160x80], bias+ReLU -> sbuf (same-wave LDS is in-order).
  for (int nt = 0; nt < 5; ++nt) {
    v8f acc = {};
#pragma unroll
    for (int kc = 0; kc < 5; ++kc) {
      v16h af = load_a16(Hh + kc * 32, CPAD, lane);
      v16h bf = load_bT(sW2 + nt * 16 * CPAD + kc * 32, CPAD, lane);
      acc = wmma16(af, bf, acc);
    }
    int col = nt * 16 + n;
    if (col < H_) {
      float bias = bc2[col];
#pragma unroll
      for (int v = 0; v < 8; ++v) {
        float x = acc[v] + bias;
        if (x < 0.f) x = 0.f;
        sbuf[(g0 + v + mb) * SPAD + col] = (h16)x;
      }
    }
  }
}

// ---------------- Kernel 4: mean/max pooling over L ----------------

__global__ __launch_bounds__(160) void k_pool(const h16* __restrict__ sbuf,
                                              float* __restrict__ vec) {
  const int b = blockIdx.x, t = threadIdx.x;
  if (t >= 140) return;
  const int sen = t / H_, h = t % H_;
  const h16* p = sbuf + ((long)sen * B_ * L_ + (long)b * L_) * SPAD + h;
  float s = 0.f, m = -1e30f;
  for (int l = 0; l < L_; ++l) {
    float x = (float)p[(long)l * SPAD];
    s += x; m = fmaxf(m, x);
  }
  vec[b * 280 + sen * 140 + h]      = s * (1.f / L_);
  vec[b * 280 + sen * 140 + 70 + h] = m;
}

// ---------------- Kernel 5: afterpool head (BN+FC+ReLU x2, final FC) ----------------

__global__ __launch_bounds__(288) void k_head(
    const float* __restrict__ vec,
    const float* g1, const float* be1, const float* m1, const float* v1,
    const float* Wa1, const float* ba1,
    const float* g2, const float* be2, const float* m2, const float* v2,
    const float* Wa2, const float* ba2,
    const float* Wfc, const float* bfc, float* __restrict__ out) {
  __shared__ float x[280], y[280], z[140];
  const int b = blockIdx.x, t = threadIdx.x;
  if (t < 280)
    x[t] = (vec[b * 280 + t] - m1[t]) * rsqrtf(v1[t] + 1e-5f) * g1[t] + be1[t];
  __syncthreads();
  if (t < 280) {
    float s = ba1[t];
    for (int k = 0; k < 280; ++k) s += x[k] * Wa1[t * 280 + k];
    s = fmaxf(s, 0.f);
    y[t] = (s - m2[t]) * rsqrtf(v2[t] + 1e-5f) * g2[t] + be2[t];
  }
  __syncthreads();
  if (t < 140) {
    float s = ba2[t];
    for (int k = 0; k < 280; ++k) s += y[k] * Wa2[t * 280 + k];
    z[t] = fmaxf(s, 0.f);
  }
  __syncthreads();
  if (t < 2) {
    float s = bfc[t];
    for (int k = 0; k < 140; ++k) s += z[k] * Wfc[t * 140 + k];
    out[b * 2 + t] = s;
  }
}

// ---------------- Launch ----------------

extern "C" void kernel_launch(void* const* d_in, const int* in_sizes, int n_in,
                              void* d_out, int out_size, void* d_ws, size_t ws_size,
                              hipStream_t stream) {
  const int*   sen1 = (const int*)d_in[0];
  const int*   sen2 = (const int*)d_in[1];
  const float* emb  = (const float*)d_in[2];
  const float* Wp   = (const float*)d_in[3];
  const float* bp   = (const float*)d_in[4];
  const float* Wc1  = (const float*)d_in[5];
  const float* bc1  = (const float*)d_in[6];
  const float* Wc2  = (const float*)d_in[7];
  const float* bc2  = (const float*)d_in[8];
  const float* g1   = (const float*)d_in[9];
  const float* be1  = (const float*)d_in[10];
  const float* m1   = (const float*)d_in[11];
  const float* v1   = (const float*)d_in[12];
  const float* Wa1  = (const float*)d_in[13];
  const float* ba1  = (const float*)d_in[14];
  const float* g2   = (const float*)d_in[15];
  const float* be2  = (const float*)d_in[16];
  const float* m2   = (const float*)d_in[17];
  const float* v2   = (const float*)d_in[18];
  const float* Wa2  = (const float*)d_in[19];
  const float* ba2  = (const float*)d_in[20];
  const float* Wfc  = (const float*)d_in[21];
  const float* bfc  = (const float*)d_in[22];
  float* out = (float*)d_out;

  char* ws = (char*)d_ws;
  size_t off = 0;
  auto alloc = [&](size_t bytes) { char* p = ws + off; off = (off + bytes + 255) & ~(size_t)255; return p; };
  h16* wWp  = (h16*)alloc((size_t)80 * PPAD * 2);
  h16* wWc1 = (h16*)alloc((size_t)144 * CPAD * 2);
  h16* wWc2 = (h16*)alloc((size_t)80 * CPAD * 2);
  h16* cbuf = (h16*)alloc((size_t)NTOK * CPAD * 2);   // ~160 MB: [p | ctx | pad] f16
  h16* sbuf = (h16*)alloc((size_t)NTOK * SPAD * 2);   // ~84 MB
  float* vec = (float*)alloc((size_t)B_ * 280 * 4);

  k_prep<<<(144 * CPAD + 255) / 256, 256, 0, stream>>>(Wp, Wc1, Wc2, wWp, wWc1, wWc2);
  k_embed_proj<<<NTOK / 128, 256, 0, stream>>>(sen1, sen2, emb, wWp, bp, cbuf);

  const size_t attn_lds = (size_t)(2 * 256 * PPAD + 256 * 256) * 2 + 4 * 256 * 4; // ~228KB
  k_attn<<<B_, 512, attn_lds, stream>>>(cbuf);

  const size_t mlp_lds = (size_t)(144 * CPAD + 80 * CPAD + 16 * 16 * CPAD) * 2;   // ~150KB
  k_mlp<<<NTOK / 128, 256, mlp_lds, stream>>>(cbuf, wWc1, wWc2, bc1, bc2, sbuf);

  k_pool<<<B_, 160, 0, stream>>>(sbuf, vec);
  k_head<<<B_, 288, 0, stream>>>(vec, g1, be1, m1, v1, Wa1, ba1,
                                 g2, be2, m2, v2, Wa2, ba2, Wfc, bfc, out);
}
